// GAT_11081015624040
// MI455X (gfx1250) — matrix operand order repaired and checked by reading, not compiled, not run on hardware
//
#include <hip/hip_runtime.h>

// ---------------- constants (match reference) ----------------
#define NN      50000
#define EE      800000
#define ETOT    (EE + NN)       // edges + self loops
#define F_IN    128
#define HID     64
#define HEADS   4
#define HC1     (HEADS * HID)   // 256
#define HID2    32
#define NCLS    40

typedef __bf16 bf16;
typedef bf16  v16bf __attribute__((ext_vector_type(16)));
typedef bf16  v8bf  __attribute__((ext_vector_type(8)));
typedef float v8f   __attribute__((ext_vector_type(8)));

// fp32 -> bf16 bits, round-to-nearest-even
__device__ __forceinline__ unsigned short f2bfbits(float f) {
    unsigned u = __float_as_uint(f);
    return (unsigned short)((u + 0x7fffu + ((u >> 16) & 1u)) >> 16);
}

// order-preserving float<->uint for atomicMax-based segment max
__device__ __forceinline__ unsigned ford(float f) {
    unsigned u = __float_as_uint(f);
    return (u & 0x80000000u) ? ~u : (u | 0x80000000u);
}
__device__ __forceinline__ float funord(unsigned u) {
    unsigned v = (u & 0x80000000u) ? (u & 0x7fffffffu) : ~u;
    return __uint_as_float(v);
}

__device__ __forceinline__ float lrelu(float v, float s) { return v > 0.f ? v : s * v; }

// ---------------------------------------------------------------------------
// Packing kernels: fp32 -> bf16 (row-major copy, and transposed weight copy)
// ---------------------------------------------------------------------------
__global__ void k_pack_bf16(const float* __restrict__ in, unsigned short* __restrict__ out,
                            long long n) {
    const long long i = (long long)blockIdx.x * blockDim.x + threadIdx.x;
    if (i < n) out[i] = f2bfbits(in[i]);
}

// Bt[n*K + k] = bf16(B[k*Nc + n])   (B row-major KxNc  ->  Bt row-major NcxK)
__global__ void k_pack_bt(const float* __restrict__ B, unsigned short* __restrict__ Bt,
                          int K, int Nc) {
    const int i = blockIdx.x * blockDim.x + threadIdx.x;
    if (i >= K * Nc) return;
    const int n = i / K, k = i % K;
    Bt[i] = f2bfbits(B[(size_t)k * Nc + n]);
}

// ---------------------------------------------------------------------------
// WMMA bf16 GEMM, pre-packed operands.
//   A  : bf16 row-major [M,K]          (K multiple of 32)
//   Bt : bf16 row-major [Nc,K]         (i.e. B transposed; Nc multiple of 16)
//   C  : f32 or bf16 row-major [M,Nc]
// One wave per 16x16 C tile; K templated so the loop fully unrolls into K/32
// v_wmma_f32_16x16x32_bf16 instructions.  ISA 7.12.2 fragment layouts:
//   A frag : lane(group g,row lr) holds K = k0+g*8+e (e<8), k0+16+g*8+(e-8)
//   B frag : lane(col,group g)    holds K = k0+g*8+e (e<8), k0+16+g*8+(e-8)
//   D frag : VGPR p -> row g*8+p, col lane%16
// ---------------------------------------------------------------------------
template <int K, bool OUT_BF16, bool ACT>
__global__ void k_gemm_wmma(const unsigned short* __restrict__ Ab,
                            const unsigned short* __restrict__ Btb,
                            const float* __restrict__ bias,
                            void* __restrict__ Cout, int M, int Nc, float slope) {
    const int lane = threadIdx.x & 31;
    const int g    = lane >> 4;
    const int lr   = lane & 15;
    const int mt   = blockIdx.y;
    const int nt   = blockIdx.x;

    int arow = mt * 16 + lr;
    if (arow >= M) arow = M - 1;            // M is a multiple of 16 here; safety only
    const int bcol = nt * 16 + lr;

    const v8bf* __restrict__ Ar = (const v8bf*)(Ab  + (size_t)arow * K);
    const v8bf* __restrict__ Br = (const v8bf*)(Btb + (size_t)bcol * K);
    __builtin_prefetch(Ar, 0, 1);           // global_prefetch_b8
    __builtin_prefetch(Br, 0, 1);

    v8f acc = {0.f, 0.f, 0.f, 0.f, 0.f, 0.f, 0.f, 0.f};

    #pragma unroll
    for (int k0 = 0; k0 < K; k0 += 32) {
        const v8bf a0 = Ar[(k0 >> 3) + g];
        const v8bf a1 = Ar[(k0 >> 3) + 2 + g];
        const v8bf b0 = Br[(k0 >> 3) + g];
        const v8bf b1 = Br[(k0 >> 3) + 2 + g];
        const v16bf a = __builtin_shufflevector(a0, a1, 0, 1, 2, 3, 4, 5, 6, 7,
                                                8, 9, 10, 11, 12, 13, 14, 15);
        const v16bf b = __builtin_shufflevector(b0, b1, 0, 1, 2, 3, 4, 5, 6, 7,
                                                8, 9, 10, 11, 12, 13, 14, 15);
        acc = __builtin_amdgcn_wmma_f32_16x16x32_bf16(
            /*neg_a=*/false, a, /*neg_b=*/false, b,
            /*c_mod=*/(short)0, acc, /*reuse_a=*/false, /*reuse_b=*/false);
    }

    #pragma unroll
    for (int p = 0; p < 8; ++p) {
        const int r = mt * 16 + g * 8 + p;
        if (r < M) {
            float v = acc[p] + (bias ? bias[bcol] : 0.f);
            if (ACT) v = lrelu(v, slope);
            if (OUT_BF16)
                ((unsigned short*)Cout)[(size_t)r * Nc + bcol] = f2bfbits(v);
            else
                ((float*)Cout)[(size_t)r * Nc + bcol] = v;
        }
    }
}

// ---------------------------------------------------------------------------
// a_s[n,h] = dot(h[n,h,:], att_src[h,:]);  a_d likewise
// ---------------------------------------------------------------------------
__global__ void k_attdot(const float* __restrict__ h, const float* __restrict__ att_s,
                         const float* __restrict__ att_d, float* __restrict__ as_,
                         float* __restrict__ ad_, int Nrows, int H, int C) {
    const int t = blockIdx.x * blockDim.x + threadIdx.x;
    if (t >= Nrows * H) return;
    const int n = t / H, hh = t % H;
    const float* hp = h + (size_t)n * H * C + (size_t)hh * C;
    const float* sp = att_s + hh * C;
    const float* dp = att_d + hh * C;
    float s = 0.f, d = 0.f;
    for (int c = 0; c < C; ++c) { const float v = hp[c]; s += v * sp[c]; d += v * dp[c]; }
    as_[t] = s; ad_[t] = d;
}

__device__ __forceinline__ void edge_sd(const int* __restrict__ ei, long long eid,
                                        int& s, int& d) {
    if (eid < EE) { s = ei[eid]; d = ei[EE + eid]; }
    else          { s = d = (int)(eid - EE); }
}

// pass 1: segment max of leaky_relu(a_s[src]+a_d[dst], 0.2) over dst
__global__ void k_edge_max(const int* __restrict__ ei, int H,
                           const float* __restrict__ as_, const float* __restrict__ ad_,
                           unsigned* __restrict__ m) {
    const long long t = (long long)blockIdx.x * blockDim.x + threadIdx.x;
    if (t >= (long long)ETOT * H) return;
    const int h = (int)(t % H);
    int s, d; edge_sd(ei, t / H, s, d);
    const float e = lrelu(as_[s * H + h] + ad_[d * H + h], 0.2f);
    atomicMax(&m[d * H + h], ford(e));
}

// pass 2: ex = exp(e - m[dst]); denom[dst] += ex
__global__ void k_edge_expsum(const int* __restrict__ ei, int H,
                              const float* __restrict__ as_, const float* __restrict__ ad_,
                              const unsigned* __restrict__ m, float* __restrict__ ex,
                              float* __restrict__ den) {
    const long long t = (long long)blockIdx.x * blockDim.x + threadIdx.x;
    if (t >= (long long)ETOT * H) return;
    const int h = (int)(t % H);
    int s, d; edge_sd(ei, t / H, s, d);
    const float e  = lrelu(as_[s * H + h] + ad_[d * H + h], 0.2f);
    const float xv = __expf(e - funord(m[d * H + h]));
    ex[t] = xv;
    atomicAdd(&den[d * H + h], xv);
}

// pass 3: out[dst,h,c] += h[src,h,c] * ex/den[dst,h]
__global__ void k_edge_agg(const int* __restrict__ ei, int H, int C,
                           const float* __restrict__ hfeat, const float* __restrict__ ex,
                           const float* __restrict__ den, float* __restrict__ outp) {
    const long long idx = (long long)blockIdx.x * blockDim.x + threadIdx.x;
    if (idx >= (long long)ETOT * H * C) return;
    const int c = (int)(idx % C);
    const long long r = idx / C;
    const int h = (int)(r % H);
    int s, d; edge_sd(ei, r / H, s, d);
    const float alpha = ex[(r / H) * H + h] / den[d * H + h];
    atomicAdd(&outp[(size_t)d * H * C + (size_t)h * C + c],
              hfeat[(size_t)s * H * C + (size_t)h * C + c] * alpha);
}

// y = leaky(x + bias[c], slope), in place
__global__ void k_bias_act(float* __restrict__ x, const float* __restrict__ bias,
                           int Nrows, int C, float slope) {
    const long long idx = (long long)blockIdx.x * blockDim.x + threadIdx.x;
    if (idx >= (long long)Nrows * C) return;
    const int c = (int)(idx % C);
    x[idx] = lrelu(x[idx] + bias[c], slope);
}

// per-feature mean / rsqrt(var+eps), one block per feature
__global__ void k_bn_stats(const float* __restrict__ x, int Nrows, int C,
                           float* __restrict__ mu, float* __restrict__ rv) {
    __shared__ float ssum[256], ssq[256];
    const int f = blockIdx.x, tid = threadIdx.x;
    float s = 0.f, q = 0.f;
    for (int r = tid; r < Nrows; r += blockDim.x) {
        const float v = x[(size_t)r * C + f]; s += v; q += v * v;
    }
    ssum[tid] = s; ssq[tid] = q; __syncthreads();
    for (int off = 128; off > 0; off >>= 1) {
        if (tid < off) { ssum[tid] += ssum[tid + off]; ssq[tid] += ssq[tid + off]; }
        __syncthreads();
    }
    if (tid == 0) {
        const float m = ssum[0] / Nrows;
        const float var = ssq[0] / Nrows - m * m;
        mu[f] = m; rv[f] = rsqrtf(var + 1e-5f);
    }
}

// normalize in place (f32 out)
__global__ void k_bn_apply(float* __restrict__ x, const float* __restrict__ mu,
                           const float* __restrict__ rv, const float* __restrict__ gm,
                           const float* __restrict__ bt, int Nrows, int C) {
    const long long idx = (long long)blockIdx.x * blockDim.x + threadIdx.x;
    if (idx >= (long long)Nrows * C) return;
    const int c = (int)(idx % C);
    x[idx] = (x[idx] - mu[c]) * rv[c] * gm[c] + bt[c];
}

// normalize, emit bf16 (feeds next WMMA GEMM directly)
__global__ void k_bn_apply_bf16(const float* __restrict__ x, const float* __restrict__ mu,
                                const float* __restrict__ rv, const float* __restrict__ gm,
                                const float* __restrict__ bt,
                                unsigned short* __restrict__ y, int Nrows, int C) {
    const long long idx = (long long)blockIdx.x * blockDim.x + threadIdx.x;
    if (idx >= (long long)Nrows * C) return;
    const int c = (int)(idx % C);
    y[idx] = f2bfbits((x[idx] - mu[c]) * rv[c] * gm[c] + bt[c]);
}

// logits = h @ Wl + bl; log_softmax;  one thread per node
__global__ void k_logits_lsm(const float* __restrict__ h, const float* __restrict__ Wl,
                             const float* __restrict__ bl, float* __restrict__ outp,
                             int Nrows) {
    const int n = blockIdx.x * blockDim.x + threadIdx.x;
    if (n >= Nrows) return;
    float lg[NCLS];
    float mx = -3.0e38f;
    const float* hp = h + (size_t)n * HID2;
    for (int c = 0; c < NCLS; ++c) {
        float a = bl[c];
        for (int k = 0; k < HID2; ++k) a += hp[k] * Wl[k * NCLS + c];
        lg[c] = a; if (a > mx) mx = a;
    }
    float se = 0.f;
    for (int c = 0; c < NCLS; ++c) se += __expf(lg[c] - mx);
    const float lse = mx + __logf(se);
    for (int c = 0; c < NCLS; ++c) outp[(size_t)n * NCLS + c] = lg[c] - lse;
}

// ---------------------------------------------------------------------------
static inline long long cdiv(long long a, long long b) { return (a + b - 1) / b; }

extern "C" void kernel_launch(void* const* d_in, const int* in_sizes, int n_in,
                              void* d_out, int out_size, void* d_ws, size_t ws_size,
                              hipStream_t stream) {
    (void)in_sizes; (void)n_in; (void)out_size; (void)ws_size;
    const float* x    = (const float*)d_in[0];
    const int*   ei   = (const int*)  d_in[1];   // [2,E] src row then dst row
    /* d_in[2] edge_weight: unused by reference */
    const float* Wf   = (const float*)d_in[3];
    const float* bf_  = (const float*)d_in[4];
    const float* W1   = (const float*)d_in[5];
    const float* aS1  = (const float*)d_in[6];
    const float* aD1  = (const float*)d_in[7];
    const float* b1   = (const float*)d_in[8];
    const float* W2   = (const float*)d_in[9];
    const float* aS2  = (const float*)d_in[10];
    const float* aD2  = (const float*)d_in[11];
    const float* b2   = (const float*)d_in[12];
    const float* g1   = (const float*)d_in[13];
    const float* be1  = (const float*)d_in[14];
    const float* g2   = (const float*)d_in[15];
    const float* be2  = (const float*)d_in[16];
    const float* Wl   = (const float*)d_in[17];
    const float* bl   = (const float*)d_in[18];
    float* outp = (float*)d_out;

    // ---- workspace layout (256B aligned, aggressive reuse) ----
    char* ws = (char*)d_ws;
    size_t o = 0;
    auto take = [&](size_t bytes) { size_t p = o; o = (o + bytes + 255) & ~(size_t)255; return p; };
    const size_t o_h1   = take((size_t)NN * HC1 * 4);       // 51.2 MB (multi-recycled)
    const size_t o_agg1 = take((size_t)NN * HC1 * 4);       // 51.2 MB (holds xb/h0b early)
    const size_t o_ex1  = take((size_t)ETOT * HEADS * 4);   // 13.6 MB
    const size_t o_as1  = take((size_t)NN * HEADS * 4);
    const size_t o_ad1  = take((size_t)NN * HEADS * 4);
    const size_t o_m1   = take((size_t)NN * HEADS * 4);
    const size_t o_den1 = take((size_t)NN * HEADS * 4);
    const size_t o_stat = take(4096);                        // mu1/rv1/mu2/rv2
    const size_t o_wfb  = take((size_t)HID  * F_IN * 2);     // Wf^T  bf16
    const size_t o_w1b  = take((size_t)HC1  * HID  * 2);     // W1^T  bf16
    const size_t o_w2b  = take((size_t)HID2 * HC1  * 2);     // W2^T  bf16

    float*    h1   = (float*)(ws + o_h1);
    float*    agg1 = (float*)(ws + o_agg1);
    float*    ex1  = (float*)(ws + o_ex1);
    float*    as1  = (float*)(ws + o_as1);
    float*    ad1  = (float*)(ws + o_ad1);
    unsigned* m1   = (unsigned*)(ws + o_m1);
    float*    den1 = (float*)(ws + o_den1);
    float*    mu1  = (float*)(ws + o_stat);
    float*    rv1  = mu1 + 256;
    float*    mu2  = rv1 + 256;
    float*    rv2  = mu2 + 64;
    unsigned short* Wfb = (unsigned short*)(ws + o_wfb);
    unsigned short* W1b = (unsigned short*)(ws + o_w1b);
    unsigned short* W2b = (unsigned short*)(ws + o_w2b);

    // early-phase tenants of the agg1 region (dead before agg1's memset)
    unsigned short* xb  = (unsigned short*)(ws + o_agg1);                         // 12.8 MB
    unsigned short* h0b = (unsigned short*)(ws + o_agg1 + (size_t)NN * F_IN * 2); //  6.4 MB
    // late-phase tenants of the h1 region (h1 dead after conv-1 aggregation)
    unsigned short* bn1b = (unsigned short*)(ws + o_h1);                          // 25.6 MB
    float* h2   = (float*)(ws + o_h1 + (size_t)NN * HC1 * 2);                     //  6.4 MB
    float* agg2 = h2 + (size_t)NN * HID2;                                         //  6.4 MB
    float* ex2  = agg2 + (size_t)NN * HID2;                                       //  3.4 MB
    // conv-2 scalars recycle conv-1 scalar regions
    float*    as2  = as1;
    float*    ad2  = ad1;
    unsigned* m2   = m1;
    float*    den2 = den1;

    const int MT = NN / 16;   // 3125, exact

    // 0) pack operands to bf16 (weights transposed for contiguous B-fragments)
    k_pack_bf16<<<(int)cdiv((long long)NN * F_IN, 256), 256, 0, stream>>>(x, xb, (long long)NN * F_IN);
    k_pack_bt  <<<(int)cdiv(F_IN * HID,  256), 256, 0, stream>>>(Wf, Wfb, F_IN, HID);
    k_pack_bt  <<<(int)cdiv(HID  * HC1,  256), 256, 0, stream>>>(W1, W1b, HID,  HC1);
    k_pack_bt  <<<(int)cdiv(HC1  * HID2, 256), 256, 0, stream>>>(W2, W2b, HC1,  HID2);

    // 1) h0 = leaky_0.01(x @ Wf + bf), emitted as bf16 (only feeds GEMM-2)
    k_gemm_wmma<F_IN, true, true><<<dim3(HID / 16, MT), 32, 0, stream>>>(
        xb, Wfb, bf_, h0b, NN, HID, 0.01f);
    // 2) h1 = h0 @ W1 (f32; bias comes after aggregation)
    k_gemm_wmma<HID, false, false><<<dim3(HC1 / 16, MT), 32, 0, stream>>>(
        h0b, W1b, nullptr, h1, NN, HC1, 0.f);
    // 3) attention dots for conv-1
    k_attdot<<<(int)cdiv((long long)NN * HEADS, 256), 256, 0, stream>>>(
        h1, aS1, aD1, as1, ad1, NN, HEADS, HID);
    // 4) segment softmax + aggregate (conv-1)
    hipMemsetAsync(m1,   0, (size_t)NN * HEADS * 4, stream);
    hipMemsetAsync(den1, 0, (size_t)NN * HEADS * 4, stream);
    hipMemsetAsync(agg1, 0, (size_t)NN * HC1   * 4, stream);
    {
        const long long teh = (long long)ETOT * HEADS;
        k_edge_max   <<<(int)cdiv(teh, 256), 256, 0, stream>>>(ei, HEADS, as1, ad1, m1);
        k_edge_expsum<<<(int)cdiv(teh, 256), 256, 0, stream>>>(ei, HEADS, as1, ad1, m1, ex1, den1);
        const long long tehc = (long long)ETOT * HEADS * HID;
        k_edge_agg   <<<(int)cdiv(tehc, 256), 256, 0, stream>>>(ei, HEADS, HID, h1, ex1, den1, agg1);
    }
    // 5) + b1, leaky_0.01, BatchNorm1 (bf16 output straight into GEMM-3)
    k_bias_act<<<(int)cdiv((long long)NN * HC1, 256), 256, 0, stream>>>(agg1, b1, NN, HC1, 0.01f);
    k_bn_stats<<<HC1, 256, 0, stream>>>(agg1, NN, HC1, mu1, rv1);
    k_bn_apply_bf16<<<(int)cdiv((long long)NN * HC1, 256), 256, 0, stream>>>(
        agg1, mu1, rv1, g1, be1, bn1b, NN, HC1);
    // 6) h2 = bn1 @ W2
    k_gemm_wmma<HC1, false, false><<<dim3(HID2 / 16, MT), 32, 0, stream>>>(
        bn1b, W2b, nullptr, h2, NN, HID2, 0.f);
    // 7) attention + segment softmax + aggregate (conv-2, 1 head)
    k_attdot<<<(int)cdiv((long long)NN, 256), 256, 0, stream>>>(h2, aS2, aD2, as2, ad2, NN, 1, HID2);
    hipMemsetAsync(m2,   0, (size_t)NN * 4, stream);
    hipMemsetAsync(den2, 0, (size_t)NN * 4, stream);
    hipMemsetAsync(agg2, 0, (size_t)NN * HID2 * 4, stream);
    {
        const long long teh = (long long)ETOT;
        k_edge_max   <<<(int)cdiv(teh, 256), 256, 0, stream>>>(ei, 1, as2, ad2, m2);
        k_edge_expsum<<<(int)cdiv(teh, 256), 256, 0, stream>>>(ei, 1, as2, ad2, m2, ex2, den2);
        const long long tehc = (long long)ETOT * HID2;
        k_edge_agg   <<<(int)cdiv(tehc, 256), 256, 0, stream>>>(ei, 1, HID2, h2, ex2, den2, agg2);
    }
    // 8) + b2, leaky_0.01, BatchNorm2 (f32 in place; feeds scalar logits kernel)
    k_bias_act<<<(int)cdiv((long long)NN * HID2, 256), 256, 0, stream>>>(agg2, b2, NN, HID2, 0.01f);
    k_bn_stats<<<HID2, 256, 0, stream>>>(agg2, NN, HID2, mu2, rv2);
    k_bn_apply<<<(int)cdiv((long long)NN * HID2, 256), 256, 0, stream>>>(
        agg2, mu2, rv2, g2, be2, NN, HID2);
    // 9) logits + log_softmax
    k_logits_lsm<<<(int)cdiv((long long)NN, 128), 128, 0, stream>>>(agg2, Wl, bl, outp, NN);
}
